// ACMConv_88802743812568
// MI455X (gfx1250) — matrix operand order repaired and checked by reading, not compiled, not run on hardware
//
#include <hip/hip_runtime.h>
#include <math.h>

#define D 128
#define LDSS 136   // padded LDS row stride in halves (16B-aligned rows, bank-conflict free)

typedef _Float16 half16_t __attribute__((ext_vector_type(16)));
typedef _Float16 half8_t  __attribute__((ext_vector_type(8)));
typedef float    float8_t __attribute__((ext_vector_type(8)));

// ---------------------------------------------------------------- degree ----
__global__ __launch_bounds__(256) void acm_deg(const int* __restrict__ row, int E,
                                               float* __restrict__ deg) {
    int e = blockIdx.x * 256 + threadIdx.x;
    if (e < E)
        __hip_atomic_fetch_add(&deg[row[e]], 1.0f, __ATOMIC_RELAXED, __HIP_MEMORY_SCOPE_AGENT);
}

__global__ __launch_bounds__(256) void acm_dinv(float* __restrict__ deg, int N) {
    int i = blockIdx.x * 256 + threadIdx.x;
    if (i < N) deg[i] = rsqrtf(deg[i] + 1.0f);   // +1 self loop; always >= 1
}

// ------------------------------------------------- fused 3x GEMM + gate -----
// Stage a 128x128 fp32 matrix -> f16 LDS tile (stride LDSS), coalesced.
__device__ __forceinline__ void stage_f16(const float* __restrict__ src,
                                          _Float16* __restrict__ dst,
                                          int t, int rowLimit, int node0b) {
    #pragma unroll
    for (int it = 0; it < 16; ++it) {
        int r = (t >> 5) + (it << 3);    // 8 rows per pass
        int c = (t & 31) << 2;           // 4 consecutive floats
        float4 v = make_float4(0.f, 0.f, 0.f, 0.f);
        if (node0b + r < rowLimit) v = *(const float4*)(src + (size_t)(node0b + r) * D + c);
        _Float16* d = dst + r * LDSS + c;
        d[0] = (_Float16)v.x; d[1] = (_Float16)v.y;
        d[2] = (_Float16)v.z; d[3] = (_Float16)v.w;
    }
}

// One 128-row x-tile times one 128x128 weight (as W^T), WMMA f32<-f16.
__device__ __forceinline__ void gemm_one(const _Float16* __restrict__ wbuf,
                                         const float* __restrict__ bb,
                                         float* __restrict__ O,
                                         const half16_t afrag[4],
                                         int lane, int node0, int N) {
    const int m  = lane & 15;
    const int hi = (lane >> 4) & 1;
    if (node0 >= N) return;                  // wave-uniform
    const bool full = (node0 + 16) <= N;     // wave-uniform fast path

    // preload the 8 per-tile biases so loads overlap compute
    float biasv[8];
    #pragma unroll
    for (int c = 0; c < 8; ++c) biasv[c] = bb[c * 16 + m];

    const int rbase = node0 + hi * 8;
    #pragma unroll
    for (int c = 0; c < 8; ++c) {            // 8 output-column tiles (unrolled -> pipelined)
        int o = c * 16 + m;
        float8_t acc;
        #pragma unroll
        for (int i = 0; i < 8; ++i) acc[i] = biasv[c];
        #pragma unroll
        for (int ks = 0; ks < 4; ++ks) {     // K = 128 in 4 steps of 32
            // B = W^T: lane (col o) supplies K = kb2..kb2+7 then kb2+16..kb2+23 from wbuf row o
            int kb2 = ks * 32 + hi * 8;
            half8_t b0 = *(const half8_t*)(wbuf + o * LDSS + kb2);
            half8_t b1 = *(const half8_t*)(wbuf + o * LDSS + kb2 + 16);
            half16_t bfrag;
            #pragma unroll
            for (int h = 0; h < 8; ++h) { bfrag[h] = b0[h]; bfrag[h + 8] = b1[h]; }
            acc = __builtin_amdgcn_wmma_f32_16x16x32_f16(
                false, afrag[ks], false, bfrag, (short)0, acc, false, false);
        }
        // D layout: VGPR r -> M = r + hi*8 ; lane m -> column o
        if (full) {
            #pragma unroll
            for (int r = 0; r < 8; ++r)
                O[(size_t)(rbase + r) * D + o] = acc[r];
        } else {
            #pragma unroll
            for (int r = 0; r < 8; ++r) {
                int gr = rbase + r;
                if (gr < N) O[(size_t)gr * D + o] = acc[r];
            }
        }
    }
}

// One block = 128 rows of x; 8 waves, each wave owns 16 rows.
__global__ __launch_bounds__(256) void acm_gemm3(
    const float* __restrict__ x,
    const float* __restrict__ Wlow,  const float* __restrict__ blow,
    const float* __restrict__ Whigh, const float* __restrict__ bhigh,
    const float* __restrict__ Wid,   const float* __restrict__ bid,
    const float* __restrict__ Wg,    const float* __restrict__ bg,
    float* __restrict__ xlow, float* __restrict__ xhigh, float* __restrict__ xid,
    float* __restrict__ gate, int N)
{
    __shared__ __align__(16) _Float16 xs  [128 * LDSS];   // 34.8 KB x tile (f16)
    __shared__ __align__(16) _Float16 wbuf[128 * LDSS];   // 34.8 KB staged weight (f16)

    const int t      = threadIdx.x;
    const int node0b = blockIdx.x * 128;
    const int lane   = t & 31;
    const int wave   = t >> 5;
    const int node0  = node0b + wave * 16;

    stage_f16(x, xs, t, N, node0b);          // x tile (zero-padded OOB rows)
    stage_f16(Wlow, wbuf, t, 1 << 30, 0);    // first weight
    __syncthreads();

    // ---- A fragments (ISA 16-bit A 16x32 layout, per 32-K step):
    // lanes 0-15: K 0-7 then 16-23 ; lanes 16-31: K 8-15 then 24-31
    const int m    = lane & 15;
    const int hi   = (lane >> 4) & 1;
    const int rowm = wave * 16 + m;
    half16_t afrag[4];
    #pragma unroll
    for (int ks = 0; ks < 4; ++ks) {
        int koff = ks * 32 + hi * 8;
        half8_t a0 = *(const half8_t*)(xs + rowm * LDSS + koff);
        half8_t a1 = *(const half8_t*)(xs + rowm * LDSS + koff + 16);
        #pragma unroll
        for (int h = 0; h < 8; ++h) { afrag[ks][h] = a0[h]; afrag[ks][h + 8] = a1[h]; }
    }

    gemm_one(wbuf, blow, xlow, afrag, lane, node0, N);
    __syncthreads();
    stage_f16(Whigh, wbuf, t, 1 << 30, 0);
    __syncthreads();
    gemm_one(wbuf, bhigh, xhigh, afrag, lane, node0, N);
    __syncthreads();
    stage_f16(Wid, wbuf, t, 1 << 30, 0);
    __syncthreads();
    gemm_one(wbuf, bid, xid, afrag, lane, node0, N);

    // ---- gate logits from the x LDS tile (3x128 GEMV, vectorized) ----
    for (int tt = t; tt < 128 * 3; tt += 256) {
        int r = tt / 3, j = tt % 3;
        int gr = node0b + r;
        if (gr < N) {
            float s = bg[j];
            const _Float16* xr = xs + r * LDSS;
            const float*    wg = Wg + j * D;
            #pragma unroll
            for (int k8 = 0; k8 < D; k8 += 8) {
                half8_t xv = *(const half8_t*)(xr + k8);
                float4 w0 = *(const float4*)(wg + k8);
                float4 w1 = *(const float4*)(wg + k8 + 4);
                s += (float)xv[0] * w0.x + (float)xv[1] * w0.y
                   + (float)xv[2] * w0.z + (float)xv[3] * w0.w
                   + (float)xv[4] * w1.x + (float)xv[5] * w1.y
                   + (float)xv[6] * w1.z + (float)xv[7] * w1.w;
            }
            gate[(size_t)gr * 3 + j] = s;
        }
    }
}

// ------------------------------------------- edge gather-scale-scatter ------
// 32 threads per edge; each thread covers 4 channels (float4 gather, 8 atomics).
__global__ __launch_bounds__(256) void acm_scatter(
    const int* __restrict__ row, const int* __restrict__ col, int E, int N,
    const float* __restrict__ dinv,
    const float* __restrict__ xlow, const float* __restrict__ xhigh,
    float* __restrict__ agg_low, float* __restrict__ agg_hl)
{
    long long g = (long long)blockIdx.x * 256 + threadIdx.x;
    int e  = (int)(g >> 5);
    int c4 = ((int)g & 31) << 2;
    int EA = E + N;
    if (e >= EA) return;
    int r, c;
    if (e < E) { r = row[e]; c = col[e]; }
    else       { r = e - E;  c = r;      }           // self loop
    float nrm = dinv[r] * dinv[c];

    const float4 vl = *(const float4*)(xlow  + (size_t)r * D + c4);
    const float4 vh = *(const float4*)(xhigh + (size_t)r * D + c4);
    float* al = agg_low + (size_t)c * D + c4;
    float* ah = agg_hl  + (size_t)c * D + c4;
    __hip_atomic_fetch_add(&al[0], nrm * vl.x, __ATOMIC_RELAXED, __HIP_MEMORY_SCOPE_AGENT);
    __hip_atomic_fetch_add(&al[1], nrm * vl.y, __ATOMIC_RELAXED, __HIP_MEMORY_SCOPE_AGENT);
    __hip_atomic_fetch_add(&al[2], nrm * vl.z, __ATOMIC_RELAXED, __HIP_MEMORY_SCOPE_AGENT);
    __hip_atomic_fetch_add(&al[3], nrm * vl.w, __ATOMIC_RELAXED, __HIP_MEMORY_SCOPE_AGENT);
    __hip_atomic_fetch_add(&ah[0], nrm * vh.x, __ATOMIC_RELAXED, __HIP_MEMORY_SCOPE_AGENT);
    __hip_atomic_fetch_add(&ah[1], nrm * vh.y, __ATOMIC_RELAXED, __HIP_MEMORY_SCOPE_AGENT);
    __hip_atomic_fetch_add(&ah[2], nrm * vh.z, __ATOMIC_RELAXED, __HIP_MEMORY_SCOPE_AGENT);
    __hip_atomic_fetch_add(&ah[3], nrm * vh.w, __ATOMIC_RELAXED, __HIP_MEMORY_SCOPE_AGENT);
}

// --------------------------------------------- softmax gate + combine -------
__global__ __launch_bounds__(256) void acm_final(
    const float* __restrict__ gate, const float* __restrict__ agg_low,
    const float* __restrict__ agg_hl, const float* __restrict__ xhigh,
    const float* __restrict__ xid, float* __restrict__ out, int N)
{
    long long g = (long long)blockIdx.x * 256 + threadIdx.x;
    int i  = (int)(g >> 5);
    int c4 = ((int)g & 31) << 2;
    if (i >= N) return;

    float l0 = gate[(size_t)i * 3 + 0];
    float l1 = gate[(size_t)i * 3 + 1];
    float l2 = gate[(size_t)i * 3 + 2];
    float mx = fmaxf(l0, fmaxf(l1, l2));
    float e0 = __expf(l0 - mx), e1 = __expf(l1 - mx), e2 = __expf(l2 - mx);
    float inv = 1.0f / (e0 + e1 + e2);
    float g0 = e0 * inv, g1 = e1 * inv, g2 = e2 * inv;

    size_t off = (size_t)i * D + c4;
    float4 al = *(const float4*)(agg_low + off);
    float4 ah = *(const float4*)(agg_hl  + off);
    float4 xh = *(const float4*)(xhigh   + off);
    float4 xi = *(const float4*)(xid     + off);
    float4 o;
    o.x = g0 * al.x + g1 * (xh.x - ah.x) + g2 * xi.x;
    o.y = g0 * al.y + g1 * (xh.y - ah.y) + g2 * xi.y;
    o.z = g0 * al.z + g1 * (xh.z - ah.z) + g2 * xi.z;
    o.w = g0 * al.w + g1 * (xh.w - ah.w) + g2 * xi.w;
    *(float4*)(out + off) = o;
}

// ---------------------------------------------------------------------------
extern "C" void kernel_launch(void* const* d_in, const int* in_sizes, int n_in,
                              void* d_out, int out_size, void* d_ws, size_t ws_size,
                              hipStream_t stream) {
    (void)n_in; (void)out_size; (void)ws_size;
    const float* x     = (const float*)d_in[0];
    const int*   ei    = (const int*)  d_in[1];
    const float* Wlow  = (const float*)d_in[2];
    const float* blow  = (const float*)d_in[3];
    const float* Whigh = (const float*)d_in[4];
    const float* bhigh = (const float*)d_in[5];
    const float* Wid   = (const float*)d_in[6];
    const float* bid   = (const float*)d_in[7];
    const float* Wg    = (const float*)d_in[8];
    const float* bg    = (const float*)d_in[9];
    float* out = (float*)d_out;

    const int N = in_sizes[0] / D;
    const int E = in_sizes[1] / 2;
    const int* erow = ei;
    const int* ecol = ei + E;

    // workspace layout (floats): deg | xlow | xhigh | xid | agg_low | agg_hl | gate
    float* ws      = (float*)d_ws;
    float* deg     = ws;                              // N (becomes dinv in place)
    float* xlow    = ws + N;                          // N*D
    float* xhigh   = xlow    + (size_t)N * D;         // N*D
    float* xid     = xhigh   + (size_t)N * D;         // N*D
    float* agg_low = xid     + (size_t)N * D;         // N*D
    float* agg_hl  = agg_low + (size_t)N * D;         // N*D
    float* gate    = agg_hl  + (size_t)N * D;         // N*3

    hipMemsetAsync(deg, 0, (size_t)N * sizeof(float), stream);
    hipMemsetAsync(agg_low, 0, (size_t)N * D * 2 * sizeof(float), stream);

    acm_deg <<<(E + 255) / 256, 256, 0, stream>>>(erow, E, deg);
    acm_dinv<<<(N + 255) / 256, 256, 0, stream>>>(deg, N);
    acm_gemm3<<<(N + 127) / 128, 256, 0, stream>>>(x, Wlow, blow, Whigh, bhigh,
                                                   Wid, bid, Wg, bg,
                                                   xlow, xhigh, xid, gate, N);
    long long sthreads = (long long)(E + N) * 32;
    acm_scatter<<<(unsigned)((sthreads + 255) / 256), 256, 0, stream>>>(
        erow, ecol, E, N, deg, xlow, xhigh, agg_low, agg_hl);
    long long fthreads = (long long)N * 32;
    acm_final<<<(unsigned)((fthreads + 255) / 256), 256, 0, stream>>>(
        gate, agg_low, agg_hl, xhigh, xid, out, N);
}
